// HybridQLSTM_65481071402093
// MI455X (gfx1250) — matrix operand order repaired and checked by reading, not compiled, not run on hardware
//
#include <hip/hip_runtime.h>
#include <math.h>

typedef __attribute__((ext_vector_type(16))) _Float16 v16h;
typedef __attribute__((ext_vector_type(8)))  _Float16 v8h;
typedef __attribute__((ext_vector_type(8)))  float    v8f;

#define T_STEPS 1024
#define BATCH   64
#define DIN     256
#define HID     256
#define KTOT    512      // D + H
#define NWG     16       // one WGP-resident workgroup per 16-column H slice
#define HSLICE  16
#define THREADS 512      // 16 waves (wave32)

// ---------------- workspace layout ----------------
// [0)        Wcvt  f16 [4][256][512]   1 MiB   (gate-major, row = h, col = k over [x|h])
// [1 MiB)    bias  f32 [4][256]        4 KiB
// [+4K)      hbuf  f16 [2][64][256]    64 KiB  (double-buffered recurrent state)
// [+64K)     ctr   u32                 barrier counter
static constexpr size_t WS_W    = 0;
static constexpr size_t WS_BIAS = (size_t)1 << 20;
static constexpr size_t WS_H    = WS_BIAS + 4096;
static constexpr size_t WS_CTR  = WS_H + 2ull * BATCH * HID * sizeof(_Float16);

// ---------------- prep: f32 -> f16 weights, zero state ----------------
__global__ void lstm_prep(const float* __restrict__ Wf, const float* __restrict__ bf,
                          const float* __restrict__ Wi, const float* __restrict__ bi,
                          const float* __restrict__ Wg, const float* __restrict__ bg,
                          const float* __restrict__ Wo, const float* __restrict__ bo,
                          _Float16* __restrict__ wcvt, float* __restrict__ bcvt,
                          _Float16* __restrict__ hbuf, unsigned* __restrict__ ctr) {
  const float* W[4]  = {Wf, Wi, Wg, Wo};
  const float* Bv[4] = {bf, bi, bg, bo};
  int idx = blockIdx.x * blockDim.x + threadIdx.x;
  const int perG = HID * KTOT;
  if (idx < 4 * perG) {
    int g = idx / perG, r = idx - g * perG;
    wcvt[idx] = (_Float16)W[g][r];
  }
  if (idx < 4 * HID) {
    int g = idx >> 8;
    bcvt[idx] = Bv[g][idx & (HID - 1)];
  }
  if (idx < 2 * BATCH * HID) hbuf[idx] = (_Float16)0.f;
  if (idx == 0) *ctr = 0u;
}

// ---------------- agent-scope grid barrier (16 resident WGs) ----------------
__device__ __forceinline__ void grid_barrier(unsigned* ctr, unsigned target) {
  __threadfence();                 // each thread's global h-stores become agent-visible
  __syncthreads();
  if (threadIdx.x == 0) {
    __hip_atomic_fetch_add(ctr, 1u, __ATOMIC_RELEASE, __HIP_MEMORY_SCOPE_AGENT);
    while (__hip_atomic_load(ctr, __ATOMIC_ACQUIRE, __HIP_MEMORY_SCOPE_AGENT) < target) {
      __builtin_amdgcn_s_sleep(1);
    }
  }
  __syncthreads();
}

// ---------------- persistent recurrent kernel ----------------
__global__ __launch_bounds__(THREADS, 1)
void lstm_recurrent(const float* __restrict__ x,        // [T,B,D] f32
                    const _Float16* __restrict__ wcvt,  // [4][H][KTOT] f16
                    const float* __restrict__ bcvt,     // [4][H] f32
                    _Float16* __restrict__ hbuf,        // [2][B][H] f16
                    unsigned* __restrict__ ctr,
                    float* __restrict__ out)            // stacked[T,B,H] + hx[B,H] + cx[B,H]
{
  extern __shared__ char smem[];
  _Float16* comb = (_Float16*)smem;                 // [64][512] f16  64 KiB  ([x_t | h])
  _Float16* wlds = (_Float16*)(smem + 65536);       // [4][16][512]   64 KiB  weight slice
  float*    gact = (float*)(smem + 131072);         // [4][64][16]    16 KiB  post-activation gates
  float*    cbuf = (float*)(smem + 147456);         // [64][16]        4 KiB  cell state slice
  float*    blds = (float*)(smem + 151552);         // [64]           256 B   bias slice

  const int tid   = threadIdx.x;
  const int lane  = tid & 31;
  const int wv    = tid >> 5;        // 0..15 : one 16x16 output tile per wave
  const int gate  = wv & 3;          // f,i,g,o
  const int mtile = wv >> 2;         // batch tile 0..3
  const int h0    = blockIdx.x * HSLICE;
  const int nlo   = lane & 15;
  const int khalf = lane >> 4;       // 0: lanes 0-15, 1: lanes 16-31

  // ---- pin this WG's weight slice (all 4 gates, 16 H-cols, K=512) in LDS ----
  for (int i = tid; i < 4 * HSLICE * KTOT; i += THREADS) {
    int g = i / (HSLICE * KTOT);
    int r = i - g * (HSLICE * KTOT);
    int n = r / KTOT;
    int k = r - n * KTOT;
    wlds[i] = wcvt[((size_t)g * HID + (h0 + n)) * KTOT + k];
  }
  for (int i = tid; i < 4 * HSLICE; i += THREADS)
    blds[i] = bcvt[(i >> 4) * HID + h0 + (i & 15)];
  for (int i = tid; i < BATCH * HSLICE; i += THREADS) cbuf[i] = 0.f;
  __syncthreads();

  // ---- preload B operands into registers, held across ALL timesteps ----
  // B (32x16 f16): lane holds col N=lane&15; halfs 0-7 = K 0..7 (+8 if upper half-wave),
  // halfs 8-15 = K 16..23 (+8) of W row (h0+N) -> two ds_read_b128 each.
  v16h Breg[16];
  {
    const _Float16* wb = wlds + ((gate * HSLICE + nlo) * KTOT) + (khalf << 3);
#pragma unroll
    for (int ks = 0; ks < 16; ++ks) {
      const _Float16* p = wb + ks * 32;
      v8h lo = *(const v8h*)p;
      v8h hi = *(const v8h*)(p + 16);
      v16h b;
#pragma unroll
      for (int j = 0; j < 8; ++j) { b[j] = lo[j]; b[j + 8] = hi[j]; }
      Breg[ks] = b;
    }
  }

  // A (16x32 f16): lane holds row M=lane&15; halfs 0-7 = K 0..7 (+8 if upper half-wave),
  // halfs 8-15 = K 16..23 (+8) -> two ds_read_b128 per WMMA.
  const _Float16* arow = comb + ((mtile * 16 + nlo) * KTOT) + (khalf << 3);
  const float bias = blds[gate * HSLICE + nlo];

  for (int t = 0; t < T_STEPS; ++t) {
    const _Float16* hRead  = hbuf + (size_t)(t & 1) * (BATCH * HID);
    _Float16*       hWrite = hbuf + (size_t)((t + 1) & 1) * (BATCH * HID);

    // ---- phase 1: stage comb = [x_t | h_t] into LDS ----
    const float* xt = x + (size_t)t * (BATCH * DIN);
    for (int i = tid; i < BATCH * DIN; i += THREADS) {
      int b = i >> 8, k = i & (DIN - 1);
      comb[b * KTOT + k] = (_Float16)xt[i];
    }
    for (int i = tid; i < (BATCH * HID) / 8; i += THREADS) {
      int b = i >> 5, k8 = i & 31;
      *(v8h*)(comb + b * KTOT + DIN + k8 * 8) = *(const v8h*)(hRead + b * HID + k8 * 8);
    }
    if (t + 1 < T_STEPS) {  // hide next step's only HBM traffic
      const char* nx = (const char*)(x + (size_t)(t + 1) * (BATCH * DIN));
      __builtin_prefetch(nx + tid * 128, 0, 1);
    }
    __syncthreads();

    // ---- phase 2: gate GEMM, one 16x16 tile per wave, K=512, dual acc chains ----
    v8f acc0 = {}; v8f acc1 = {};
#pragma unroll
    for (int ks = 0; ks < 16; ks += 2) {
      {
        const _Float16* pa = arow + ks * 32;
        v8h a0 = *(const v8h*)pa;
        v8h a1 = *(const v8h*)(pa + 16);
        v16h A;
#pragma unroll
        for (int j = 0; j < 8; ++j) { A[j] = a0[j]; A[j + 8] = a1[j]; }
        acc0 = __builtin_amdgcn_wmma_f32_16x16x32_f16(false, A, false, Breg[ks],
                                                      (short)0, acc0, false, false);
      }
      {
        const _Float16* pa = arow + (ks + 1) * 32;
        v8h a0 = *(const v8h*)pa;
        v8h a1 = *(const v8h*)(pa + 16);
        v16h A;
#pragma unroll
        for (int j = 0; j < 8; ++j) { A[j] = a0[j]; A[j + 8] = a1[j]; }
        acc1 = __builtin_amdgcn_wmma_f32_16x16x32_f16(false, A, false, Breg[ks + 1],
                                                      (short)0, acc1, false, false);
      }
    }
    // D layout: lanes 0-15 -> M = r, lanes 16-31 -> M = 8+r; N = lane&15
    const int mrow0 = mtile * 16 + khalf * 8;
#pragma unroll
    for (int r = 0; r < 8; ++r) {
      float v = acc0[r] + acc1[r] + bias;
      v = (gate == 2) ? tanhf(v) : (1.f / (1.f + expf(-v)));
      gact[(gate * BATCH + (mrow0 + r)) * HSLICE + nlo] = v;
    }
    __syncthreads();

    // ---- phase 3: local elementwise c/h update for this slice ----
#pragma unroll
    for (int e = 0; e < 2; ++e) {
      int i = tid + e * THREADS;            // 0..1023 over [64][16]
      int b = i >> 4, n = i & 15;
      float f  = gact[(0 * BATCH + b) * HSLICE + n];
      float ii = gact[(1 * BATCH + b) * HSLICE + n];
      float gg = gact[(2 * BATCH + b) * HSLICE + n];
      float oo = gact[(3 * BATCH + b) * HSLICE + n];
      float c  = f * cbuf[i] + ii * gg;
      cbuf[i]  = c;
      float h  = oo * tanhf(c);
      hWrite[b * HID + h0 + n] = (_Float16)h;
      out[(size_t)t * (BATCH * HID) + b * HID + h0 + n] = h;
      if (t == T_STEPS - 1)
        out[(size_t)T_STEPS * (BATCH * HID) + b * HID + h0 + n] = h;  // hx
    }

    // ---- one barrier per step (double-buffered h makes this sufficient) ----
    grid_barrier(ctr, (unsigned)(t + 1) * NWG);
  }

  // ---- cx from LDS cell state ----
  for (int i = tid; i < BATCH * HSLICE; i += THREADS) {
    int b = i >> 4, n = i & 15;
    out[(size_t)(T_STEPS + 1) * (BATCH * HID) + b * HID + h0 + n] = cbuf[i];
  }
}

extern "C" void kernel_launch(void* const* d_in, const int* in_sizes, int n_in,
                              void* d_out, int out_size, void* d_ws, size_t ws_size,
                              hipStream_t stream) {
  (void)in_sizes; (void)n_in; (void)out_size; (void)ws_size;
  const float* x  = (const float*)d_in[0];
  const float* Wf = (const float*)d_in[1];
  const float* bf = (const float*)d_in[2];
  const float* Wi = (const float*)d_in[3];
  const float* bi = (const float*)d_in[4];
  const float* Wg = (const float*)d_in[5];
  const float* bg = (const float*)d_in[6];
  const float* Wo = (const float*)d_in[7];
  const float* bo = (const float*)d_in[8];

  char* ws = (char*)d_ws;
  _Float16* wcvt = (_Float16*)(ws + WS_W);
  float*    bcvt = (float*)(ws + WS_BIAS);
  _Float16* hbuf = (_Float16*)(ws + WS_H);
  unsigned* ctr  = (unsigned*)(ws + WS_CTR);

  lstm_prep<<<1024, 512, 0, stream>>>(Wf, bf, Wi, bi, Wg, bg, Wo, bo,
                                      wcvt, bcvt, hbuf, ctr);

  const size_t shmem = 151552 + 256;  // comb 64K + W 64K + gact 16K + c 4K + bias
  lstm_recurrent<<<NWG, THREADS, shmem, stream>>>(x, wcvt, bcvt, hbuf, ctr,
                                                  (float*)d_out);
}
